// ParT_LHAASO_4561255268375
// MI455X (gfx1250) — matrix operand order repaired and checked by reading, not compiled, not run on hardware
//
#include <hip/hip_runtime.h>
#include <math.h>
#include <cstdint>

// ---------------------------------------------------------------------------
// ParT-style transformer forward for MI455X (gfx1250, wave32, WMMA).
// All matmuls with M>=16 go through v_wmma_f32_16x16x32_f16 (f16 in, f32 acc).
// CDNA5 async global->LDS copies (ASYNCcnt) used for conversion-free staging.
// ---------------------------------------------------------------------------

typedef _Float16 half_t;
typedef __attribute__((ext_vector_type(16))) _Float16 v16h;
typedef __attribute__((ext_vector_type(8)))  float    v8f;

#define DEV static __device__ __forceinline__

// problem constants
#define BB   4
#define NN   512
#define DD   128
#define HH   8
#define HDIM 16
#define NLAY 4

DEV float geluf(float x) { return 0.5f * x * (1.0f + erff(x * 0.7071067811865475f)); }

// --- WMMA fragment builders (per CDNA5 ISA 7.12.2 layouts, wave32) ----------
// A (16x32 f16, row-major tile, row stride ld):
//   lane L: M = L%16, g = L/16; e<8 -> K = g*8+e ; e>=8 -> K = 16+g*8+(e-8)
DEV v16h wmma_a_frag(const half_t* A, int ld, int lane) {
  int M = lane & 15, g = lane >> 4;
  const half_t* row = A + M * ld;
  v16h f;
#pragma unroll
  for (int e = 0; e < 8; ++e) f[e] = row[g * 8 + e];
#pragma unroll
  for (int e = 0; e < 8; ++e) f[8 + e] = row[16 + g * 8 + e];
  return f;
}
// B passed transposed: Bt[N][K] (row stride ld). lane L: N=L%16, e<8 -> K=g*8+e ; e>=8 -> K=16+g*8+(e-8).
DEV v16h wmma_bt_frag(const half_t* Bt, int ld, int lane) {
  int n = lane & 15, g = lane >> 4;
  const half_t* row = Bt + n * ld;
  v16h f;
#pragma unroll
  for (int e = 0; e < 16; ++e) f[e] = row[(e < 8) ? (g * 8 + e) : (16 + g * 8 + (e - 8))];
  return f;
}
DEV v8f wmma16(v16h a, v16h b, v8f c) {
  return __builtin_amdgcn_wmma_f32_16x16x32_f16(false, a, false, b, (short)0, c,
                                                false, false);
}

// --- 128-thread (4-wave) block sum ------------------------------------------
DEV float block_sum128(float v, float* sbuf) {
  int t = threadIdx.x;
#pragma unroll
  for (int off = 1; off < 32; off <<= 1) v += __shfl_xor(v, off, 32);
  __syncthreads();
  if ((t & 31) == 0) sbuf[t >> 5] = v;
  __syncthreads();
  return sbuf[0] + sbuf[1] + sbuf[2] + sbuf[3];
}

// ===========================================================================
// 1) Embedding: feat = gelu(ln(hits@embed_w+b)) + gelu(pos@pos_w+pos_b)
//    grid = B*N blocks, 128 threads (thread = output channel d)
// ===========================================================================
__global__ void embed_kernel(const float* __restrict__ hits,
                             const float* __restrict__ ew, const float* __restrict__ eb,
                             const float* __restrict__ lng, const float* __restrict__ lnb,
                             const float* __restrict__ pw, const float* __restrict__ pb,
                             float* __restrict__ feat) {
  __shared__ float sbuf[4];
  int row = blockIdx.x;
  int t = threadIdx.x;
  const float* x = hits + row * 5;
  float e = eb[t];
#pragma unroll
  for (int p = 0; p < 5; ++p) e += x[p] * ew[p * DD + t];
  float mean = block_sum128(e, sbuf) * (1.0f / DD);
  float d = e - mean;
  float var = block_sum128(d * d, sbuf) * (1.0f / DD);
  float y = geluf(d * rsqrtf(var + 1e-5f) * lng[t] + lnb[t]);
  float pos = geluf(x[0] * pw[t] + x[1] * pw[DD + t] + pb[t]);
  feat[row * DD + t] = y + pos;
}

// ===========================================================================
// 2) Pair means: mt[b][c] = mean_{i,j}|c_i - c_j| + 1e-6 for channels 2..4
//    grid = B*3 blocks, 256 threads.
//    LDS staging uses CDNA5 GLOBAL_LOAD_ASYNC_TO_LDS_B32 (ASYNCcnt path).
// ===========================================================================
__global__ void pair_mean_kernel(const float* __restrict__ hits, float* __restrict__ mt) {
  __shared__ float cs[NN];
  __shared__ float partial[8];
  int b = blockIdx.x / 3, c = blockIdx.x % 3;
  int t = threadIdx.x;
#pragma unroll
  for (int it = 0; it < 2; ++it) {
    int i = t + it * 256;
    // LDS byte offset = low 32 bits of the generic (flat) shared-memory addr
    unsigned int ldsoff = (unsigned int)(uintptr_t)(&cs[i]);
    unsigned long long ga =
        (unsigned long long)(uintptr_t)(hits + (size_t)(b * NN + i) * 5 + 2 + c);
    asm volatile("global_load_async_to_lds_b32 %0, %1, off"
                 :: "v"(ldsoff), "v"(ga) : "memory");
  }
  asm volatile("s_wait_asynccnt 0x0" ::: "memory");
  __syncthreads();
  float s = 0.f;
  for (int idx = t; idx < NN * NN; idx += 256) {
    int i = idx >> 9, j = idx & (NN - 1);
    s += fabsf(cs[i] - cs[j]);
  }
#pragma unroll
  for (int off = 1; off < 32; off <<= 1) s += __shfl_xor(s, off, 32);
  if ((t & 31) == 0) partial[t >> 5] = s;
  __syncthreads();
  if (t == 0) {
    float tot = 0.f;
#pragma unroll
    for (int w = 0; w < 8; ++w) tot += partial[w];
    mt[b * 3 + c] = tot / (float)(NN * NN) + 1e-6f;
  }
}

// ===========================================================================
// 3) Generic GEMM  C[M,N] = act(A[M,K] @ Bw[K,N] + bias)
//    f32 in -> f16 LDS staging -> v_wmma_f32_16x16x32_f16 -> f32 out.
//    Block = 128 threads (4 waves); tile 64x64; K stepped by 32.
//    Staging uses fixed trip counts: issue 8 b128 loads, then convert/store.
// ===========================================================================
__global__ void gemm_wmma(const float* __restrict__ A, const float* __restrict__ Bw,
                          const float* __restrict__ bias, float* __restrict__ C,
                          int K, int Nld, int act) {
  __shared__ half_t Ah[64 * 32];
  __shared__ half_t Bt[64 * 32];
  int m0 = blockIdx.x * 64;
  int n0 = blockIdx.y * 64;
  int t = threadIdx.x, wave = t >> 5, lane = t & 31;
  v8f acc[4] = {};
  for (int k0 = 0; k0 < K; k0 += 32) {
    float4 va[4], vb[4];
#pragma unroll
    for (int it = 0; it < 4; ++it) {
      int idx = t * 4 + it * 512;
      int r = idx >> 5, c = idx & 31;          // A tile coords
      va[it] = *(const float4*)(A + (size_t)(m0 + r) * K + k0 + c);
      int kk = idx >> 6, nn = idx & 63;        // B tile coords
      vb[it] = *(const float4*)(Bw + (size_t)(k0 + kk) * Nld + n0 + nn);
    }
#pragma unroll
    for (int it = 0; it < 4; ++it) {
      int idx = t * 4 + it * 512;
      int r = idx >> 5, c = idx & 31;
      Ah[r * 32 + c + 0] = (half_t)va[it].x;
      Ah[r * 32 + c + 1] = (half_t)va[it].y;
      Ah[r * 32 + c + 2] = (half_t)va[it].z;
      Ah[r * 32 + c + 3] = (half_t)va[it].w;
      int kk = idx >> 6, nn = idx & 63;
      Bt[(nn + 0) * 32 + kk] = (half_t)vb[it].x;
      Bt[(nn + 1) * 32 + kk] = (half_t)vb[it].y;
      Bt[(nn + 2) * 32 + kk] = (half_t)vb[it].z;
      Bt[(nn + 3) * 32 + kk] = (half_t)vb[it].w;
    }
    __syncthreads();
    v16h af = wmma_a_frag(Ah + wave * 16 * 32, 32, lane);
#pragma unroll
    for (int nt = 0; nt < 4; ++nt) {
      v16h bf = wmma_bt_frag(Bt + nt * 16 * 32, 32, lane);
      acc[nt] = wmma16(af, bf, acc[nt]);
    }
    __syncthreads();
  }
  int nl = lane & 15, g = lane >> 4;
#pragma unroll
  for (int nt = 0; nt < 4; ++nt) {
    int n = n0 + nt * 16 + nl;
    float bv = bias ? bias[n] : 0.f;
#pragma unroll
    for (int r = 0; r < 8; ++r) {
      int m = m0 + wave * 16 + r + 8 * g;
      float x = acc[nt][r] + bv;
      if (act == 1) x = geluf(x);
      C[(size_t)m * Nld + n] = x;
    }
  }
}

// ===========================================================================
// 4) Pair bias: bias[b][h][i][j] = relu(pf@pm1+b1) @ pm2[:,h] + b2[h]
//    pf = (c[i]-c[j])/mt per channel 2..4. Hidden (3->128) via VALU,
//    128->8 projection via WMMA (N padded to 16). One wave per (b,i,jtile).
// ===========================================================================
__global__ void pair_bias_kernel(const float* __restrict__ hits, const float* __restrict__ mt,
                                 const float* __restrict__ pm1_w, const float* __restrict__ pm1_b,
                                 const float* __restrict__ pm2_w, const float* __restrict__ pm2_b,
                                 float* __restrict__ biasOut) {
  __shared__ half_t hid[4][16 * 128];  // per-wave: 16 pairs x 128 hidden
  __shared__ half_t pm2t[16 * 128];    // [n][k]; n>=8 zero padding
  int t = threadIdx.x, wave = t >> 5, lane = t & 31;
#pragma unroll
  for (int it = 0; it < 16; ++it) {
    int idx = t + it * 128;
    int n = idx >> 7, k = idx & 127;
    pm2t[idx] = (n < 8) ? (half_t)pm2_w[k * HH + n] : (half_t)0.f;
  }
  int wid = blockIdx.x * 4 + wave;      // 0 .. B*N*32-1
  int b = wid >> 14;
  int i = (wid >> 5) & (NN - 1);
  int jt = wid & 31;
  const float* hb = hits + (size_t)b * NN * 5;
  float inv0 = 1.f / mt[b * 3 + 0];
  float inv1 = 1.f / mt[b * 3 + 1];
  float inv2 = 1.f / mt[b * 3 + 2];
  float ci0 = hb[i * 5 + 2], ci1 = hb[i * 5 + 3], ci2 = hb[i * 5 + 4];
  for (int m = 0; m < 16; ++m) {
    int j = jt * 16 + m;
    float p0 = (ci0 - hb[j * 5 + 2]) * inv0;
    float p1 = (ci1 - hb[j * 5 + 3]) * inv1;
    float p2 = (ci2 - hb[j * 5 + 4]) * inv2;
#pragma unroll
    for (int dd = 0; dd < 4; ++dd) {
      int d = lane + dd * 32;
      float hv = p0 * pm1_w[d] + p1 * pm1_w[DD + d] + p2 * pm1_w[2 * DD + d] + pm1_b[d];
      hid[wave][m * 128 + d] = (half_t)fmaxf(hv, 0.f);
    }
  }
  __syncthreads();
  v8f acc = {};
#pragma unroll
  for (int ks = 0; ks < 4; ++ks) {
    v16h af = wmma_a_frag(&hid[wave][ks * 32], 128, lane);
    v16h bf = wmma_bt_frag(&pm2t[ks * 32], 128, lane);
    acc = wmma16(af, bf, acc);
  }
  int n = lane & 15, g = lane >> 4;
  if (n < HH) {
    float bv = pm2_b[n];
#pragma unroll
    for (int r = 0; r < 8; ++r) {
      int m = r + 8 * g;  // pair index within j-tile
      biasOut[(((size_t)(b * HH + n) * NN) + i) * NN + jt * 16 + m] = acc[r] + bv;
    }
  }
}

// ===========================================================================
// 5) Attention (flash-style): per wave = one (b,h,query-tile of 16).
//    S = (q*scale) @ k^T via WMMA (K=16 zero-padded to 32), + bias,
//    online softmax (16-lane shfl reductions match C layout), O += P@V via WMMA.
// ===========================================================================
__global__ void attn_kernel(const float* __restrict__ qkv,
                            const float* __restrict__ biasIn,
                            float* __restrict__ attnOut) {
  __shared__ half_t qs[4][256];
  __shared__ half_t kbuf[4][256];  // [key][dim]  == Bt[N=key][K=dim]
  __shared__ half_t vtb[4][256];   // [dim][key]  == Bt[N=dim][K=key]
  __shared__ half_t ps[4][256];    // P tile [query][key]
  int t = threadIdx.x, wave = t >> 5, lane = t & 31;
  int wid = blockIdx.x * 4 + wave;  // 0..1023
  int b = wid >> 8;
  int h = (wid >> 5) & 7;
  int it0 = wid & 31;
  int i0 = it0 * 16;
  const float scale = 0.25f;  // HD^-0.5
#pragma unroll
  for (int it = 0; it < 8; ++it) {
    int q = lane + it * 32;
    int m = q >> 4, d = q & 15;
    qs[wave][q] = (half_t)(qkv[(size_t)(b * NN + i0 + m) * (3 * DD) + h * HDIM + d] * scale);
  }
  __syncthreads();
  int nl = lane & 15, g = lane >> 4;
  v16h qf;
  {
#pragma unroll
    for (int e = 0; e < 8; ++e) qf[e] = qs[wave][nl * 16 + g * 8 + e];
#pragma unroll
    for (int e = 8; e < 16; ++e) qf[e] = (half_t)0.f;
  }
  float m8[8], l8[8];
#pragma unroll
  for (int r = 0; r < 8; ++r) { m8[r] = -1e30f; l8[r] = 0.f; }
  v8f O = {};
  for (int jt = 0; jt < 32; ++jt) {
    int j0 = jt * 16;
    float kv[8], vv[8];
#pragma unroll
    for (int it = 0; it < 8; ++it) {
      int q = lane + it * 32;
      int m = q >> 4, d = q & 15;
      const float* row = qkv + (size_t)(b * NN + j0 + m) * (3 * DD) + h * HDIM + d;
      kv[it] = row[DD];
      vv[it] = row[2 * DD];
    }
#pragma unroll
    for (int it = 0; it < 8; ++it) {
      int q = lane + it * 32;
      int m = q >> 4, d = q & 15;
      kbuf[wave][m * 16 + d] = (half_t)kv[it];
      vtb[wave][d * 16 + m] = (half_t)vv[it];
    }
    __syncthreads();
    v16h kf;
#pragma unroll
    for (int e = 0; e < 16; ++e) kf[e] = (e < 8) ? kbuf[wave][nl * 16 + g * 8 + e] : (half_t)0.f;
    v8f S = {};
    S = wmma16(qf, kf, S);
    const float* bp = biasIn + (((size_t)(b * HH + h) * NN) + i0) * NN + j0;
#pragma unroll
    for (int r = 0; r < 8; ++r) {
      float sv = S[r] + bp[(size_t)(r + 8 * g) * NN + nl];
      float mx = sv;
#pragma unroll
      for (int off = 1; off < 16; off <<= 1) mx = fmaxf(mx, __shfl_xor(mx, off, 32));
      float mnew = fmaxf(m8[r], mx);
      float corr = expf(m8[r] - mnew);
      float p = expf(sv - mnew);
      float rs = p;
#pragma unroll
      for (int off = 1; off < 16; off <<= 1) rs += __shfl_xor(rs, off, 32);
      l8[r] = l8[r] * corr + rs;
      m8[r] = mnew;
      O[r] *= corr;
      ps[wave][(r + 8 * g) * 16 + nl] = (half_t)p;
    }
    __syncthreads();
    v16h pf;
    {
#pragma unroll
      for (int e = 0; e < 8; ++e) pf[e] = ps[wave][nl * 16 + g * 8 + e];
#pragma unroll
      for (int e = 8; e < 16; ++e) pf[e] = (half_t)0.f;
    }
    v16h vf;
#pragma unroll
    for (int e = 0; e < 16; ++e) vf[e] = (e < 8) ? vtb[wave][nl * 16 + g * 8 + e] : (half_t)0.f;
    O = wmma16(pf, vf, O);
    __syncthreads();
  }
#pragma unroll
  for (int r = 0; r < 8; ++r) {
    int m = r + 8 * g;
    attnOut[(size_t)(b * NN + i0 + m) * DD + h * HDIM + nl] = O[r] / l8[r];
  }
}

// ===========================================================================
// 6) LN with residual: feat = ln(feat + add) * g + b   (grid B*N, 128 thr)
// ===========================================================================
__global__ void ln_residual_kernel(float* __restrict__ feat, const float* __restrict__ add,
                                   const float* __restrict__ gg, const float* __restrict__ bb) {
  __shared__ float sbuf[4];
  int row = blockIdx.x, t = threadIdx.x;
  float x = feat[(size_t)row * DD + t] + add[(size_t)row * DD + t];
  float mean = block_sum128(x, sbuf) * (1.0f / DD);
  float d = x - mean;
  float var = block_sum128(d * d, sbuf) * (1.0f / DD);
  feat[(size_t)row * DD + t] = d * rsqrtf(var + 1e-5f) * gg[t] + bb[t];
}

// ===========================================================================
// 7) Max pool over N (mask all-true) -> hcat[b][0:128]
// ===========================================================================
__global__ void pool_kernel(const float* __restrict__ feat, float* __restrict__ hcat) {
  int b = blockIdx.x, t = threadIdx.x;
  float m = -1e30f;
  for (int i = 0; i < NN; ++i) m = fmaxf(m, feat[(size_t)(b * NN + i) * DD + t]);
  hcat[b * 256 + t] = m;
}

// ===========================================================================
// 8) Param head -> hcat[b][128:256]
// ===========================================================================
__global__ void param_head_kernel(const float* __restrict__ params,
                                  const float* __restrict__ w1, const float* __restrict__ b1,
                                  const float* __restrict__ lg, const float* __restrict__ lb,
                                  const float* __restrict__ w2, const float* __restrict__ b2,
                                  float* __restrict__ hcat) {
  __shared__ float sbuf[4];
  __shared__ float pbuf[DD];
  int b = blockIdx.x, t = threadIdx.x;
  const float* pr = params + b * 5;
  float v = b1[t];
#pragma unroll
  for (int p = 0; p < 5; ++p) v += pr[p] * w1[p * DD + t];
  v = geluf(v);
  float mean = block_sum128(v, sbuf) * (1.0f / DD);
  float d = v - mean;
  float var = block_sum128(d * d, sbuf) * (1.0f / DD);
  pbuf[t] = d * rsqrtf(var + 1e-5f) * lg[t] + lb[t];
  __syncthreads();
  float o = b2[t];
  for (int k = 0; k < DD; ++k) o += pbuf[k] * w2[k * DD + t];
  hcat[b * 256 + DD + t] = geluf(o);
}

// ===========================================================================
// 9) Classifier head: 256 -> 256 -> 128 -> 2   (grid B, 256 threads)
// ===========================================================================
__global__ void cls_kernel(const float* __restrict__ hcat,
                           const float* __restrict__ w1, const float* __restrict__ b1,
                           const float* __restrict__ w2, const float* __restrict__ b2,
                           const float* __restrict__ w3, const float* __restrict__ b3,
                           float* __restrict__ out) {
  __shared__ float h0[256], h1[256], h2[128];
  int b = blockIdx.x, t = threadIdx.x;
  h0[t] = hcat[b * 256 + t];
  __syncthreads();
  float a = b1[t];
  for (int k = 0; k < 256; ++k) a += h0[k] * w1[k * 256 + t];
  h1[t] = geluf(a);
  __syncthreads();
  if (t < 128) {
    float a2 = b2[t];
    for (int k = 0; k < 256; ++k) a2 += h1[k] * w2[k * 128 + t];
    h2[t] = geluf(a2);
  }
  __syncthreads();
  if (t < 2) {
    float a3 = b3[t];
    for (int k = 0; k < 128; ++k) a3 += h2[k] * w3[k * 2 + t];
    out[b * 2 + t] = a3;
  }
}

// ===========================================================================
// Host launch. Inputs flattened in JAX tree order (sorted dict keys):
//  0 hits, 1 mask(ignored; all-true), 2 params,
//  3 + 15*l + {0 ffn1_b,1 ffn1_w,2 ffn2_b,3 ffn2_w,4 n1_b,5 n1_g,6 n2_b,
//              7 n2_g,8 out_b,9 out_w,10 pm1_b,11 pm1_w,12 pm2_b,13 pm2_w,14 qkv_w}
//  63 cls1_b,64 cls1_w,65 cls2_b,66 cls2_w,67 cls3_b,68 cls3_w,
//  69 embed_b,70 embed_ln_b,71 embed_ln_g,72 embed_w,
//  73 pfc1_b,74 pfc1_w,75 pfc2_b,76 pfc2_w,77 pfc_ln_b,78 pfc_ln_g,
//  79 pos_b,80 pos_w
// ===========================================================================
extern "C" void kernel_launch(void* const* d_in, const int* in_sizes, int n_in,
                              void* d_out, int out_size, void* d_ws, size_t ws_size,
                              hipStream_t stream) {
  (void)in_sizes; (void)n_in; (void)out_size; (void)ws_size;
  const float* hits   = (const float*)d_in[0];
  const float* params = (const float*)d_in[2];
  auto W = [&](int i) { return (const float*)d_in[i]; };
  const int BL = 3;   // first block leaf
  const int TL = 63;  // first top-level weight leaf after blocks

  float* w = (float*)d_ws;
  float* feat    = w;                                       // B*N*D
  float* qkvbuf  = feat    + (size_t)BB * NN * DD;          // B*N*3D
  float* biasbuf = qkvbuf  + (size_t)BB * NN * 3 * DD;      // B*H*N*N (L2-resident)
  float* attnbuf = biasbuf + (size_t)BB * HH * NN * NN;     // B*N*D
  float* t1      = attnbuf + (size_t)BB * NN * DD;          // B*N*4D
  float* t2      = t1      + (size_t)BB * NN * 4 * DD;      // B*N*D
  float* mt      = t2      + (size_t)BB * NN * DD;          // B*3
  float* hcat    = mt      + 16;                            // B*256

  // embed + pair means
  embed_kernel<<<BB * NN, 128, 0, stream>>>(hits, W(TL + 9), W(TL + 6), W(TL + 8),
                                            W(TL + 7), W(TL + 17), W(TL + 16), feat);
  pair_mean_kernel<<<BB * 3, 256, 0, stream>>>(hits, mt);

  for (int l = 0; l < NLAY; ++l) {
    const float* ffn1_b = W(BL + l * 15 + 0);
    const float* ffn1_w = W(BL + l * 15 + 1);
    const float* ffn2_b = W(BL + l * 15 + 2);
    const float* ffn2_w = W(BL + l * 15 + 3);
    const float* n1_b   = W(BL + l * 15 + 4);
    const float* n1_g   = W(BL + l * 15 + 5);
    const float* n2_b   = W(BL + l * 15 + 6);
    const float* n2_g   = W(BL + l * 15 + 7);
    const float* out_b  = W(BL + l * 15 + 8);
    const float* out_w  = W(BL + l * 15 + 9);
    const float* pm1_b  = W(BL + l * 15 + 10);
    const float* pm1_w  = W(BL + l * 15 + 11);
    const float* pm2_b  = W(BL + l * 15 + 12);
    const float* pm2_w  = W(BL + l * 15 + 13);
    const float* qkv_w  = W(BL + l * 15 + 14);

    // qkv = feat @ qkv_w   (2048x128 @ 128x384)
    gemm_wmma<<<dim3(32, 6), 128, 0, stream>>>(feat, qkv_w, nullptr, qkvbuf, DD, 3 * DD, 0);
    // pair bias tensor (B,H,N,N) — stays resident in L2 (33.5MB << 192MB)
    pair_bias_kernel<<<BB * NN * 32 / 4, 128, 0, stream>>>(hits, mt, pm1_w, pm1_b,
                                                           pm2_w, pm2_b, biasbuf);
    // flash attention
    attn_kernel<<<BB * HH * 32 / 4, 128, 0, stream>>>(qkvbuf, biasbuf, attnbuf);
    // out projection + residual LN
    gemm_wmma<<<dim3(32, 2), 128, 0, stream>>>(attnbuf, out_w, out_b, t2, DD, DD, 0);
    ln_residual_kernel<<<BB * NN, 128, 0, stream>>>(feat, t2, n1_g, n1_b);
    // FFN
    gemm_wmma<<<dim3(32, 8), 128, 0, stream>>>(feat, ffn1_w, ffn1_b, t1, DD, 4 * DD, 1);
    gemm_wmma<<<dim3(32, 2), 128, 0, stream>>>(t1, ffn2_w, ffn2_b, t2, 4 * DD, DD, 0);
    ln_residual_kernel<<<BB * NN, 128, 0, stream>>>(feat, t2, n2_g, n2_b);
  }

  // pooling, param head, classifier
  pool_kernel<<<BB, 128, 0, stream>>>(feat, hcat);
  param_head_kernel<<<BB, 128, 0, stream>>>(params, W(TL + 11), W(TL + 10), W(TL + 15),
                                            W(TL + 14), W(TL + 13), W(TL + 12), hcat);
  cls_kernel<<<BB, 256, 0, stream>>>(hcat, W(TL + 1), W(TL + 0), W(TL + 3), W(TL + 2),
                                     W(TL + 5), W(TL + 4), (float*)d_out);
}
